// CrossAttention_49606872269322
// MI455X (gfx1250) — compile-verified, matrix operand order checked by reading, not run on hardware
//
#include <hip/hip_runtime.h>

// ---------------------------------------------------------------------------
// CDNA5 (gfx1250) fused cross-attention:
//   proj_kernel x3 : Q/K/V = X*W^T + b  (f32 in -> bf16 out, V transposed)
//   attn_kernel    : flash attention, v_wmma_f32_16x16x32_bf16, with
//                    double-buffered async GLOBAL->LDS staging of K/V chunks
//                    shared by the 4 waves of each workgroup.
// Shapes: B=8, SQ=SKV=2048, H=256.  SCALE = sqrt(H/2) = sqrt(128).
// ---------------------------------------------------------------------------

typedef __bf16 bf16_t;
typedef __attribute__((ext_vector_type(16))) __bf16 v16bf;
typedef __attribute__((ext_vector_type(8)))  __bf16 v8bf;
typedef __attribute__((ext_vector_type(8)))  float  v8f;
typedef __attribute__((ext_vector_type(4)))  float  v4f;

#define HID   256
#define SEQ   2048
#define NB    8
#define INV_SCALE 0.08838834764831845f  // 1/sqrt(128)

#if __has_builtin(__builtin_amdgcn_global_load_async_to_lds_b128)
#define HAVE_ASYNC_LDS 1
typedef int v4i_t __attribute__((vector_size(16)));            // int4
typedef __attribute__((address_space(1))) void*  as1_void_t;
typedef __attribute__((address_space(3))) void*  as3_void_t;
typedef __attribute__((address_space(1))) v4i_t* as1_v4i_t;
typedef __attribute__((address_space(3))) v4i_t* as3_v4i_t;
#else
#define HAVE_ASYNC_LDS 0
#endif

static __device__ inline v8f wmma_bf16(v16bf a, v16bf b, v8f c) {
  return __builtin_amdgcn_wmma_f32_16x16x32_bf16(
      /*neg_a=*/false, a, /*neg_b=*/false, b,
      /*c_mod=*/(short)0, c, /*reuse_a=*/false, /*reuse_b=*/false);
}

// Load a 16x32 bf16 A/B fragment slice for this lane: two 16B loads
// (global_load_b128 or ds_load_b128 depending on pointer origin).
static __device__ inline v16bf load_frag_bf16(const bf16_t* p) {
  union { v16bf v; v8bf h[2]; } u;
  u.h[0] = *(const v8bf*)(p);
  u.h[1] = *(const v8bf*)(p + 16);
  return u.v;
}

// Same, but source is f32: load 2x (8 floats) and convert to bf16 (RNE).
static __device__ inline v16bf load_frag_f32(const float* p0, const float* p1) {
  union { v16bf v; __bf16 e[16]; } u;
  v4f a = *(const v4f*)(p0);
  v4f b = *(const v4f*)(p0 + 4);
  v4f c = *(const v4f*)(p1);
  v4f d = *(const v4f*)(p1 + 4);
#pragma unroll
  for (int i = 0; i < 4; ++i) {
    u.e[i]      = (__bf16)a[i];
    u.e[4 + i]  = (__bf16)b[i];
    u.e[8 + i]  = (__bf16)c[i];
    u.e[12 + i] = (__bf16)d[i];
  }
  return u.v;
}

// 16-byte global -> LDS copy: async DMA (ASYNCcnt) when available.
static __device__ inline void copy16_g2l(const bf16_t* g, bf16_t* l) {
#if HAVE_ASYNC_LDS
  __builtin_amdgcn_global_load_async_to_lds_b128(
      (as1_v4i_t)(as1_void_t)(void*)g,
      (as3_v4i_t)(as3_void_t)(void*)l,
      /*offset=*/0, /*cpol=*/0);
#else
  *(v8bf*)l = *(const v8bf*)g;
#endif
}

static __device__ inline void stage_wait_and_sync() {
#if HAVE_ASYNC_LDS
#if __has_builtin(__builtin_amdgcn_s_wait_asynccnt)
  __builtin_amdgcn_s_wait_asynccnt(0);
#else
  asm volatile("s_wait_asynccnt 0" ::: "memory");
#endif
#endif
  __syncthreads();
}

// ---------------------------------------------------------------------------
// Projection: out = X @ W^T + bias   (torch Linear).  X:[rows,256] f32,
// W:[256,256] f32 row-major, out bf16.
// transpose_out==0 : out[row*256 + e]
// transpose_out==1 : out[(b*256 + e)*rows_per_batch + s]   (V^T layout)
// One wave computes a 16-row x 256-col tile. 4 waves / block.
// ---------------------------------------------------------------------------
__global__ __launch_bounds__(128)
void proj_kernel(const float* __restrict__ X, const float* __restrict__ W,
                 const float* __restrict__ bias, bf16_t* __restrict__ out,
                 int rows_total, int rows_per_batch, int transpose_out) {
  const int lane = threadIdx.x & 31;
  const int wave = threadIdx.x >> 5;
  const int tile = blockIdx.x * 4 + wave;     // 16-row tile
  const int r0   = tile * 16;
  if (r0 >= rows_total) return;               // wave-uniform
  const int m = lane & 15;
  const int h = lane >> 4;

  // A fragments: 8 d-chunks of the 16 input rows (lane -> row r0+m).
  v16bf afr[8];
  const float* xrow = X + (size_t)(r0 + m) * HID;
#pragma unroll
  for (int f = 0; f < 8; ++f) {
    const int d0 = f * 32 + 8 * h;
    afr[f] = load_frag_f32(xrow + d0, xrow + d0 + 16);
  }

  // C accumulators seeded with bias (per-column value, same in all 8 VGPRs).
  v8f acc[16];
#pragma unroll
  for (int e = 0; e < 16; ++e) {
    const float bv = bias[e * 16 + m];
    acc[e] = (v8f){bv, bv, bv, bv, bv, bv, bv, bv};
  }

  for (int f = 0; f < 8; ++f) {
    const int d0 = f * 32 + 8 * h;
#pragma unroll
    for (int e = 0; e < 16; ++e) {
      const float* wrow = W + (size_t)(e * 16 + m) * HID;   // lane -> out col
      v16bf bfr = load_frag_f32(wrow + d0, wrow + d0 + 16);
      acc[e] = wmma_bf16(afr[f], bfr, acc[e]);
    }
  }

  // Store bf16. C layout: VGPR r, lane -> row r0 + r + 8h, col e*16 + m.
#pragma unroll
  for (int e = 0; e < 16; ++e) {
#pragma unroll
    for (int r = 0; r < 8; ++r) {
      const int row = r0 + r + 8 * h;
      const int col = e * 16 + m;
      const bf16_t val = (bf16_t)acc[e][r];
      if (!transpose_out) {
        out[(size_t)row * HID + col] = val;
      } else {
        const int b = row / rows_per_batch;
        const int s = row - b * rows_per_batch;
        out[((size_t)b * HID + col) * rows_per_batch + s] = val;
      }
    }
  }
}

// ---------------------------------------------------------------------------
// Flash attention.  4 waves / block; each wave owns 16 query rows of the same
// batch (32 blocks per batch).  Per 32-key chunk the block cooperatively
// async-stages K (32x256) and V^T (256x32) into LDS (double buffered), then
// every wave computes S = Q K^T, online softmax, O += P V from LDS fragments.
// Qb:[B,SQ,256] bf16, Kb:[B,SKV,256] bf16, Vt:[B,256,SKV] bf16, Out f32.
// ---------------------------------------------------------------------------
struct __align__(16) AttnSmem {
  bf16_t K[2][32 * HID];    // 16 KB per buffer
  bf16_t V[2][HID * 32];    // 16 KB per buffer
  bf16_t P[4][16][32];      // per-wave P bounce (4 KB)
};

// Issue the async copies for one 32-key chunk (1024+1024 16B pieces over
// 128 threads: 8+8 issues each).
static __device__ inline void stage_chunk(const bf16_t* __restrict__ Kbase,
                                          const bf16_t* __restrict__ Vbase,
                                          bf16_t* ldsK, bf16_t* ldsV,
                                          int c, int tid) {
  const bf16_t* kg = Kbase + (size_t)c * HID;   // contiguous 16 KB
#pragma unroll
  for (int j = 0; j < 8; ++j) {
    const int p = tid + j * 128;                // 16B piece index
    copy16_g2l(kg + p * 8, ldsK + p * 8);
  }
#pragma unroll
  for (int j = 0; j < 8; ++j) {
    const int p   = tid + j * 128;              // 1024 pieces: 256 rows x 4
    const int row = p >> 2;
    const int q   = p & 3;
    copy16_g2l(Vbase + (size_t)row * SEQ + c + q * 8,
               ldsV + row * 32 + q * 8);
  }
}

__global__ __launch_bounds__(128)
void attn_kernel(const bf16_t* __restrict__ Qb, const bf16_t* __restrict__ Kb,
                 const bf16_t* __restrict__ Vt, float* __restrict__ Out) {
  __shared__ AttnSmem smem;
  const int tid  = threadIdx.x;
  const int lane = tid & 31;
  const int wave = tid >> 5;
  const int tile = blockIdx.x * 4 + wave;     // 0 .. B*SQ/16-1 = 1023
  const int b    = tile >> 7;                 // / (SQ/16 = 128)
  const int q0   = (tile & 127) << 4;
  const int m = lane & 15;
  const int h = lane >> 4;

  // Preload Q A-fragments (16 rows x 256 d = 8 fragments).
  v16bf qf[8];
  const bf16_t* qrow = Qb + ((size_t)b * SEQ + q0 + m) * HID;
#pragma unroll
  for (int f = 0; f < 8; ++f) {
    const int d0 = f * 32 + 8 * h;
    qf[f] = load_frag_bf16(qrow + d0);
  }

  const v8f vzero = (v8f){0.f, 0.f, 0.f, 0.f, 0.f, 0.f, 0.f, 0.f};
  v8f acc[16];                                // O accumulator: 16 rows x 256 d
#pragma unroll
  for (int e = 0; e < 16; ++e) acc[e] = vzero;

  float mst[8], lst[8];                       // stats for row r + 8h
#pragma unroll
  for (int r = 0; r < 8; ++r) { mst[r] = -__builtin_inff(); lst[r] = 0.f; }

  const bf16_t* Kbase = Kb + (size_t)b * SEQ * HID;
  const bf16_t* Vbase = Vt + (size_t)b * HID * SEQ;

  // Prologue: stage chunk 0 into buffer 0.
  stage_chunk(Kbase, Vbase, smem.K[0], smem.V[0], 0, tid);
  stage_wait_and_sync();

  int buf = 0;
  for (int c = 0; c < SEQ; c += 32, buf ^= 1) {
    // Kick off the DMA for the next chunk (overlaps with the WMMAs below).
    if (c + 32 < SEQ)
      stage_chunk(Kbase, Vbase, smem.K[buf ^ 1], smem.V[buf ^ 1], c + 32, tid);

    const bf16_t* Kl = smem.K[buf];
    const bf16_t* Vl = smem.V[buf];

    // ---- scores: two 16x16 tiles (keys c..c+15, c+16..c+31) ----
    v8f s0 = vzero, s1 = vzero;
#pragma unroll
    for (int f = 0; f < 8; ++f) {
      const int d0 = f * 32 + 8 * h;
      s0 = wmma_bf16(qf[f], load_frag_bf16(Kl + (size_t)(m)      * HID + d0), s0);
      s1 = wmma_bf16(qf[f], load_frag_bf16(Kl + (size_t)(16 + m) * HID + d0), s1);
    }

    float p0[8], p1[8], cmax[8];
#pragma unroll
    for (int r = 0; r < 8; ++r) {
      p0[r] = s0[r] * INV_SCALE;
      p1[r] = s1[r] * INV_SCALE;
      cmax[r] = fmaxf(p0[r], p1[r]);
    }
    // Row reductions stay inside each 16-lane half (row = r + 8h).
#pragma unroll
    for (int r = 0; r < 8; ++r) {
      float v = cmax[r];
      v = fmaxf(v, __shfl_xor(v, 1, 32));
      v = fmaxf(v, __shfl_xor(v, 2, 32));
      v = fmaxf(v, __shfl_xor(v, 4, 32));
      v = fmaxf(v, __shfl_xor(v, 8, 32));
      cmax[r] = v;
    }
    float alpha[8];
#pragma unroll
    for (int r = 0; r < 8; ++r) {
      const float mnew = fmaxf(mst[r], cmax[r]);
      alpha[r] = __expf(mst[r] - mnew);
      mst[r] = mnew;
      p0[r] = __expf(p0[r] - mnew);
      p1[r] = __expf(p1[r] - mnew);
      float s = p0[r] + p1[r];
      s += __shfl_xor(s, 1, 32);
      s += __shfl_xor(s, 2, 32);
      s += __shfl_xor(s, 4, 32);
      s += __shfl_xor(s, 8, 32);
      lst[r] = lst[r] * alpha[r] + s;
    }
    // Rescale running O.
#pragma unroll
    for (int e = 0; e < 16; ++e)
#pragma unroll
      for (int r = 0; r < 8; ++r) acc[e][r] = acc[e][r] * alpha[r];

    // ---- P (C layout, f32) -> LDS -> A-fragment (bf16) ----
#pragma unroll
    for (int r = 0; r < 8; ++r) {
      smem.P[wave][r + 8 * h][m]      = (bf16_t)p0[r];
      smem.P[wave][r + 8 * h][16 + m] = (bf16_t)p1[r];
    }
    __builtin_amdgcn_wave_barrier();
    asm volatile("s_wait_dscnt 0" ::: "memory");
    union { v16bf v; v8bf hh[2]; } pu;
    pu.hh[0] = *(const v8bf*)&smem.P[wave][m][8 * h];
    pu.hh[1] = *(const v8bf*)&smem.P[wave][m][16 + 8 * h];
    __builtin_amdgcn_wave_barrier();

    // ---- O += P @ V  (B-frags from staged V^T rows: contiguous keys) ----
#pragma unroll
    for (int e = 0; e < 16; ++e) {
      acc[e] = wmma_bf16(pu.v, load_frag_bf16(Vl + (e * 16 + m) * 32 + 8 * h),
                         acc[e]);
    }

    // Wait for the prefetch DMA + make the buffer visible to all waves; the
    // barrier also guarantees nobody overwrites a buffer still being read.
    stage_wait_and_sync();
  }

  // Epilogue: divide by l, store f32 (coalesced across lanes per VGPR).
  float* obase = Out + (size_t)b * SEQ * HID;
#pragma unroll
  for (int r = 0; r < 8; ++r) {
    const float inv_l = 1.0f / lst[r];
    const int row = q0 + r + 8 * h;
#pragma unroll
    for (int e = 0; e < 16; ++e) {
      obase[(size_t)row * HID + e * 16 + m] = acc[e][r] * inv_l;
    }
  }
}

// ---------------------------------------------------------------------------
extern "C" void kernel_launch(void* const* d_in, const int* in_sizes, int n_in,
                              void* d_out, int out_size, void* d_ws, size_t ws_size,
                              hipStream_t stream) {
  const float* query = (const float*)d_in[0];
  const float* key_  = (const float*)d_in[1];
  const float* value = (const float*)d_in[2];
  const float* Wq = (const float*)d_in[3];
  const float* bq = (const float*)d_in[4];
  const float* Wk = (const float*)d_in[5];
  const float* bk = (const float*)d_in[6];
  const float* Wv = (const float*)d_in[7];
  const float* bv = (const float*)d_in[8];
  float* out = (float*)d_out;

  // Workspace: Qb | Kb | Vt, each B*SEQ*HID bf16 = 8 MB (24 MB total).
  bf16_t* Qb = (bf16_t*)d_ws;
  bf16_t* Kb = Qb + (size_t)NB * SEQ * HID;
  bf16_t* Vt = Kb + (size_t)NB * SEQ * HID;

  const int rows = NB * SEQ;            // 16384
  const int tiles = rows / 16;          // 1024
  dim3 blk(128);                        // 4 waves
  dim3 grd(tiles / 4);                  // 256 blocks

  proj_kernel<<<grd, blk, 0, stream>>>(query, Wq, bq, Qb, rows, SEQ, 0);
  proj_kernel<<<grd, blk, 0, stream>>>(key_,  Wk, bk, Kb, rows, SEQ, 0);
  proj_kernel<<<grd, blk, 0, stream>>>(value, Wv, bv, Vt, rows, SEQ, 1);
  attn_kernel<<<grd, blk, 0, stream>>>(Qb, Kb, Vt, out);
}
